// HypMLR_70901320122487
// MI455X (gfx1250) — matrix operand order repaired and checked by reading, not compiled, not run on hardware
//
#include <hip/hip_runtime.h>
#include <hip/hip_fp16.h>
#include <math.h>

typedef _Float16 v8h  __attribute__((ext_vector_type(8)));
typedef _Float16 v16h __attribute__((ext_vector_type(16)));
typedef float    v8f  __attribute__((ext_vector_type(8)));

// ---------------- problem sizes (from setup_inputs) ----------------
#define NB     4096          // batch
#define H1     28
#define C1     32            // conv1 out channels
#define H2     14            // after pool1
#define C2     64            // conv2 out channels
#define H3     7             // after pool2
#define K1P    32            // conv1 GEMM K (9 taps padded to 32)
#define K2     288           // 9 taps * 32 ch  (conv2 GEMM K)
#define KFC    3136          // 7*7*64          (fc GEMM K)
#define DOUT   32            // fc out dim
#define KCLS   10            // classes

static __device__ inline v8h zero8h() {
    v8h r;
#pragma unroll
    for (int i = 0; i < 8; ++i) r[i] = (_Float16)0.0f;
    return r;
}

// =====================================================================
// Weight repack kernels (fp32 -> f16, GEMM-friendly row-major N x K)
// =====================================================================
__global__ __launch_bounds__(256) void repack_w1(
    const float* __restrict__ w,    // (32,1,3,3)
    _Float16* __restrict__ wp)      // (32, 32): K = tap (0..8), 9..31 = 0
{
    const int i = blockIdx.x * 256 + threadIdx.x;    // 32*32 = 1024
    if (i >= C1 * K1P) return;
    const int o = i / K1P, k = i % K1P;
    wp[i] = (k < 9) ? (_Float16)w[o * 9 + k] : (_Float16)0.0f;
}

__global__ __launch_bounds__(256) void repack_w2(
    const float* __restrict__ w,    // (64,32,3,3) OIHW
    _Float16* __restrict__ wp)      // (64, 288): K = tap*32 + c
{
    const int i = blockIdx.x * 256 + threadIdx.x;    // 64*288 = 18432
    if (i >= C2 * K2) return;
    const int o = i / K2, k = i % K2;
    const int tap = k / C1, c = k % C1;
    wp[i] = (_Float16)w[(o * C1 + c) * 9 + tap];
}

__global__ __launch_bounds__(256) void repack_wfc(
    const float* __restrict__ w,    // (32, 3136): K NCHW = c*49 + pos
    _Float16* __restrict__ wp)      // (32, 3136): K NHWC = pos*64 + c
{
    const int i = blockIdx.x * 256 + threadIdx.x;    // 32*3136 = 100352
    if (i >= DOUT * KFC) return;
    const int o = i / KFC, k = i % KFC;              // k is NHWC index
    const int pos = k / C2, c = k % C2;
    wp[i] = (_Float16)w[o * KFC + c * (H3 * H3) + pos];
}

// =====================================================================
// Kernel 1: conv1 (1->32, 3x3 pad1) as implicit GEMM with WMMA f16
// (K = 9 taps zero-padded to 32), fused relu + 2x2 maxpool.
//   M = N*28*28 flattened as (n, PY, PX, quadrant) -> 16 M rows / wave
//   N = 32 channels -> 2 WMMA column tiles, B loop-invariant
// A layout (16-bit): lanes 0-15 hold K=0..7 (taps 0-7), lanes 16-31
// hold K=8..15 (tap 8 + zeros); K=16..31 halves are all zero.
// Output NHWC f16: (N,14,14,32).
// =====================================================================
__global__ __launch_bounds__(256) void conv1_wmma_relu_pool(
    const float* __restrict__ x,        // (N,1,28,28)
    const _Float16* __restrict__ wp,    // (32,32) f16, K-padded
    const float* __restrict__ bias,     // (32)
    _Float16* __restrict__ out)         // (N,14,14,32) NHWC f16
{
    const int wid  = blockIdx.x * 8 + (threadIdx.x >> 5);
    const int lane = threadIdx.x & 31;
    const int col  = lane & 15;
    const int half = lane >> 4;

    // decode A-row: m = pooled*4 + quadrant over (n, PY, PX) in 14x14
    const int m      = wid * 16 + col;
    const int pooled = m >> 2, q = m & 3;
    const int n   = pooled / (H2 * H2);
    const int rem = pooled % (H2 * H2);
    const int PY  = rem / H2, PX = rem % H2;
    const int y   = PY * 2 + (q >> 1);
    const int xx  = PX * 2 + (q & 1);
    const float* xin = x + (size_t)n * (H1 * H1);

    // gather this lane's A halves: k = half*8 + j, valid taps are k < 9
    v8h lo;
#pragma unroll
    for (int j = 0; j < 8; ++j) {
        const int k = half * 8 + j;
        float v = 0.0f;
        if (k < 9) {
            const int dy = k / 3 - 1, dx = k % 3 - 1;
            const int iy = y + dy, ix = xx + dx;
            if ((unsigned)iy < (unsigned)H1 && (unsigned)ix < (unsigned)H1)
                v = xin[iy * H1 + ix];
        }
        lo[j] = (_Float16)v;
    }
    const v8h  hi = zero8h();                       // K=16..31 all zero
    const v16h a  = __builtin_shufflevector(lo, hi, 0,1,2,3,4,5,6,7,8,9,10,11,12,13,14,15);

    // B fragments (loop-invariant, 2 KB total): lane = column col, K = {half*8..+8} U {16+half*8..+8}
    const _Float16* wr = wp + col * K1P + half * 8;
    const v16h b0 = __builtin_shufflevector(*(const v8h*)(wr), *(const v8h*)(wr + 16), 0,1,2,3,4,5,6,7,8,9,10,11,12,13,14,15);
    const v16h b1 = __builtin_shufflevector(*(const v8h*)(wr + 16 * K1P), *(const v8h*)(wr + 16 * K1P + 16), 0,1,2,3,4,5,6,7,8,9,10,11,12,13,14,15);

    v8f acc[2];
#pragma unroll
    for (int t = 0; t < 2; ++t)
#pragma unroll
        for (int e = 0; e < 8; ++e) acc[t][e] = 0.0f;

    acc[0] = __builtin_amdgcn_wmma_f32_16x16x32_f16(false, a, false, b0, (short)0, acc[0], false, false);
    acc[1] = __builtin_amdgcn_wmma_f32_16x16x32_f16(false, a, false, b1, (short)0, acc[1], false, false);

    // epilogue: maxpool over quadrant groups of 4 D-rows, + bias, relu
    const int ppbase = wid * 4 + half * 2;
#pragma unroll
    for (int nt = 0; nt < 2; ++nt) {
        const int  ch = nt * 16 + col;
        const float bv = bias[ch];
        const v8f  A = acc[nt];
        float p0 = fmaxf(fmaxf(A[0], A[1]), fmaxf(A[2], A[3]));
        float p1 = fmaxf(fmaxf(A[4], A[5]), fmaxf(A[6], A[7]));
        p0 = fmaxf(p0 + bv, 0.0f);
        p1 = fmaxf(p1 + bv, 0.0f);
        out[(size_t)ppbase * C1 + ch]       = (_Float16)p0;
        out[(size_t)(ppbase + 1) * C1 + ch] = (_Float16)p1;
    }
}

// =====================================================================
// Kernel 2: conv2 (32->64) as implicit GEMM with WMMA f16, weights
// staged in LDS (36 KB) once per block, fused relu + 2x2 maxpool.
//   M = N*14*14 flattened as (n, py, px, quadrant)  -> 16 M rows / wave
//   K = 288 (9 taps x 32 ch), N = 64 (4 WMMA column tiles)
// Output NHWC f16: (N,7,7,64).
// =====================================================================
__global__ __launch_bounds__(256) void conv2_wmma_relu_pool(
    const _Float16* __restrict__ act,   // (N,14,14,32) f16
    const _Float16* __restrict__ wp,    // (64,288) f16
    const float* __restrict__ bias,     // (64)
    _Float16* __restrict__ out)         // (N,7,7,64) f16
{
    __shared__ _Float16 sw[C2 * K2];                // 36,864 B
    {
        // cooperative copy: 18432 halves = 2304 x 16B chunks, 9 per thread
        const uint4* src = (const uint4*)wp;
        uint4*       dst = (uint4*)sw;
        const int tid = threadIdx.x;
#pragma unroll
        for (int i = 0; i < 9; ++i) dst[tid + i * 256] = src[tid + i * 256];
    }
    __syncthreads();

    const int wid  = blockIdx.x * 8 + (threadIdx.x >> 5);
    const int lane = threadIdx.x & 31;
    const int col  = lane & 15;
    const int half = lane >> 4;

    const int m      = wid * 16 + col;
    const int pooled = m >> 2, q = m & 3;
    const int n   = pooled / (H3 * H3);
    const int rem = pooled % (H3 * H3);
    const int py  = rem / H3, px = rem % H3;
    const int y   = py * 2 + (q >> 1);
    const int xx  = px * 2 + (q & 1);
    const _Float16* an = act + (size_t)n * (H2 * H2 * C1);

    v8f acc[4];
#pragma unroll
    for (int t = 0; t < 4; ++t)
#pragma unroll
        for (int e = 0; e < 8; ++e) acc[t][e] = 0.0f;

#pragma unroll
    for (int tap = 0; tap < 9; ++tap) {
        const int dy = tap / 3 - 1, dx = tap % 3 - 1;
        const int iy = y + dy, ix = xx + dx;
        const bool valid = (unsigned)iy < (unsigned)H2 && (unsigned)ix < (unsigned)H2;

        // A fragment: lane holds K = {half*8..+8} U {16+half*8..+8}
        v8h lo = zero8h(), hi = zero8h();
        if (valid) {
            const _Float16* p = an + (iy * H2 + ix) * C1 + half * 8;
            lo = *(const v8h*)p;
            hi = *(const v8h*)(p + 16);
        }
        const v16h a = __builtin_shufflevector(lo, hi, 0,1,2,3,4,5,6,7,8,9,10,11,12,13,14,15);

        // B fragments from LDS: lane = column col, K = {half*8..+8} U {16+half*8..+8}
        const _Float16* wrow = sw + col * K2 + tap * C1 + half * 8;
        const v16h b0 = __builtin_shufflevector(*(const v8h*)(wrow), *(const v8h*)(wrow + 16), 0,1,2,3,4,5,6,7,8,9,10,11,12,13,14,15);
        const v16h b1 = __builtin_shufflevector(*(const v8h*)(wrow + 16 * K2), *(const v8h*)(wrow + 16 * K2 + 16), 0,1,2,3,4,5,6,7,8,9,10,11,12,13,14,15);
        const v16h b2 = __builtin_shufflevector(*(const v8h*)(wrow + 32 * K2), *(const v8h*)(wrow + 32 * K2 + 16), 0,1,2,3,4,5,6,7,8,9,10,11,12,13,14,15);
        const v16h b3 = __builtin_shufflevector(*(const v8h*)(wrow + 48 * K2), *(const v8h*)(wrow + 48 * K2 + 16), 0,1,2,3,4,5,6,7,8,9,10,11,12,13,14,15);

        acc[0] = __builtin_amdgcn_wmma_f32_16x16x32_f16(false, a, false, b0, (short)0, acc[0], false, false);
        acc[1] = __builtin_amdgcn_wmma_f32_16x16x32_f16(false, a, false, b1, (short)0, acc[1], false, false);
        acc[2] = __builtin_amdgcn_wmma_f32_16x16x32_f16(false, a, false, b2, (short)0, acc[2], false, false);
        acc[3] = __builtin_amdgcn_wmma_f32_16x16x32_f16(false, a, false, b3, (short)0, acc[3], false, false);
    }

    const int ppbase = wid * 4 + half * 2;
#pragma unroll
    for (int nt = 0; nt < 4; ++nt) {
        const int  ch = nt * 16 + col;
        const float bv = bias[ch];
        const v8f  A = acc[nt];
        float p0 = fmaxf(fmaxf(A[0], A[1]), fmaxf(A[2], A[3]));
        float p1 = fmaxf(fmaxf(A[4], A[5]), fmaxf(A[6], A[7]));
        p0 = fmaxf(p0 + bv, 0.0f);
        p1 = fmaxf(p1 + bv, 0.0f);
        out[(size_t)ppbase * C2 + ch]       = (_Float16)p0;
        out[(size_t)(ppbase + 1) * C2 + ch] = (_Float16)p1;
    }
}

// =====================================================================
// Kernel 3: FC (4096x3136) x (3136x32) with WMMA f16, fp32 accum+out.
// act (N,7,7,64) NHWC f16 is already the row-major (M x K) matrix.
// =====================================================================
__global__ __launch_bounds__(256) void fc_wmma(
    const _Float16* __restrict__ act,   // (4096, 3136) f16
    const _Float16* __restrict__ wfc,   // (32, 3136) f16
    const float* __restrict__ bias,     // (32)
    float* __restrict__ out)            // (4096, 32) f32
{
    const int wid  = blockIdx.x * 8 + (threadIdx.x >> 5);  // 0..255
    const int lane = threadIdx.x & 31;
    const int col  = lane & 15;
    const int half = lane >> 4;

    const _Float16* arow = act + (size_t)(wid * 16 + col) * KFC;

    v8f acc[2];
#pragma unroll
    for (int t = 0; t < 2; ++t)
#pragma unroll
        for (int e = 0; e < 8; ++e) acc[t][e] = 0.0f;

    for (int k = 0; k < KFC; k += 32) {
        const _Float16* p = arow + k + half * 8;
        const v8h lo = *(const v8h*)p;
        const v8h hi = *(const v8h*)(p + 16);
        const v16h a = __builtin_shufflevector(lo, hi, 0,1,2,3,4,5,6,7,8,9,10,11,12,13,14,15);

        const _Float16* wr = wfc + col * KFC + k + half * 8;
        const v16h b0 = __builtin_shufflevector(*(const v8h*)(wr), *(const v8h*)(wr + 16), 0,1,2,3,4,5,6,7,8,9,10,11,12,13,14,15);
        const v16h b1 = __builtin_shufflevector(*(const v8h*)(wr + 16 * KFC), *(const v8h*)(wr + 16 * KFC + 16), 0,1,2,3,4,5,6,7,8,9,10,11,12,13,14,15);

        acc[0] = __builtin_amdgcn_wmma_f32_16x16x32_f16(false, a, false, b0, (short)0, acc[0], false, false);
        acc[1] = __builtin_amdgcn_wmma_f32_16x16x32_f16(false, a, false, b1, (short)0, acc[1], false, false);
    }

#pragma unroll
    for (int nt = 0; nt < 2; ++nt) {
        const int ch = nt * 16 + col;
        const float bv = bias[ch];
#pragma unroll
        for (int r = 0; r < 8; ++r) {
            const int mr = wid * 16 + half * 8 + r;
            out[(size_t)mr * DOUT + ch] = acc[nt][r] + bv;
        }
    }
}

// =====================================================================
// Kernel 4: hyperbolic MLR head, fp32 scalar (c = 1). One thread per
// (sample, class). Faithful to reference incl. eps clamps / projection.
// =====================================================================
__global__ __launch_bounds__(256) void hyp_mlr(
    const float* __restrict__ X,    // (4096, 32)
    const float* __restrict__ W,    // (10, 32)
    const float* __restrict__ B,    // (10, 32)
    float* __restrict__ out)        // (4096, 10)
{
    const int idx = blockIdx.x * 256 + threadIdx.x;   // < 40960
    const int n = idx / KCLS, k = idx % KCLS;
    const float* xv = X + (size_t)n * DOUT;
    const float* wv = W + (size_t)k * DOUT;
    const float* bv = B + (size_t)k * DOUT;

    float x2 = 0.f, b2 = 0.f, inner = 0.f, w2 = 0.f, xw = 0.f, wb = 0.f;
#pragma unroll
    for (int d = 0; d < DOUT; ++d) {
        const float xd = xv[d], wd = wv[d], bd = bv[d];
        x2    = fmaf(xd, xd, x2);
        b2    = fmaf(bd, bd, b2);
        inner = fmaf(xd, -bd, inner);
        w2    = fmaf(wd, wd, w2);
        xw    = fmaf(xd, wd, xw);
        wb    = fmaf(wd, -bd, wb);
    }
    const float EPSv = 1e-5f;
    const float MAXN = 1.0f - 1e-5f;            // c = 1
    const float wnorm = sqrtf(w2);

    const float a_num = 1.0f + 2.0f * inner + x2;
    const float b_num = 1.0f - b2;
    const float denom = 1.0f + 2.0f * inner + x2 * b2;
    const float alpha = a_num / fmaxf(denom, EPSv);
    const float beta  = b_num / denom;          // raw denom, as in source
    float mob2 = alpha * alpha * b2 + 2.0f * alpha * beta * inner + beta * beta * x2;
    const float sq = sqrtf(mob2);
    const float normalizer = (sq > MAXN) ? (MAXN / fmaxf(sq, EPSv)) : 1.0f;
    mob2 = (sq < MAXN) ? mob2 : (MAXN * MAXN);

    const float inv_wn = 1.0f / fmaxf(wnorm, 1e-12f);
    const float hyper  = (alpha * wb + beta * xw) * inv_wn * normalizer;
    const float asinh_in = 2.0f * hyper / fmaxf(1.0f - mob2, EPSv);
    const float scaler   = 2.0f / fmaxf(1.0f - b2, EPSv);
    out[idx] = scaler * wnorm * asinhf(asinh_in);
}

// =====================================================================
// Launch
// =====================================================================
extern "C" void kernel_launch(void* const* d_in, const int* in_sizes, int n_in,
                              void* d_out, int out_size, void* d_ws, size_t ws_size,
                              hipStream_t stream) {
    const float* x    = (const float*)d_in[0];
    const float* c1w  = (const float*)d_in[1];
    const float* c1b  = (const float*)d_in[2];
    const float* c2w  = (const float*)d_in[3];
    const float* c2b  = (const float*)d_in[4];
    const float* fcw  = (const float*)d_in[5];
    const float* fcb  = (const float*)d_in[6];
    const float* hw   = (const float*)d_in[7];
    const float* hb   = (const float*)d_in[8];

    char* ws = (char*)d_ws;
    size_t off = 0;
    _Float16* act1 = (_Float16*)(ws + off); off += (size_t)NB * H2 * H2 * C1 * 2;  // 51.4 MB
    _Float16* act2 = (_Float16*)(ws + off); off += (size_t)NB * H3 * H3 * C2 * 2;  // 25.7 MB
    float*    fco  = (float*)   (ws + off); off += (size_t)NB * DOUT * 4;          // 0.5 MB
    _Float16* wp1  = (_Float16*)(ws + off); off += (size_t)C1 * K1P * 2;           // 2 KB
    _Float16* wp2  = (_Float16*)(ws + off); off += (size_t)C2 * K2 * 2;            // 36 KB
    _Float16* wfcp = (_Float16*)(ws + off); off += (size_t)DOUT * KFC * 2;         // 200 KB

    repack_w1 <<<(C1 * K1P + 255) / 256,   256, 0, stream>>>(c1w, wp1);
    repack_w2 <<<(C2 * K2 + 255) / 256,    256, 0, stream>>>(c2w, wp2);
    repack_wfc<<<(DOUT * KFC + 255) / 256, 256, 0, stream>>>(fcw, wfcp);

    // conv1: M = 4096*28*28 rows, 16 rows per wave, 8 waves per block
    conv1_wmma_relu_pool<<<(NB * H1 * H1) / 16 / 8, 256, 0, stream>>>(x, wp1, c1b, act1);

    // conv2: M = 4096*14*14 rows, 16 rows per wave, 8 waves per block
    conv2_wmma_relu_pool<<<(NB * H2 * H2) / 16 / 8, 256, 0, stream>>>(act1, wp2, c2b, act2);

    // fc: M = 4096 rows -> 256 waves -> 32 blocks
    fc_wmma<<<(NB / 16) / 8, 256, 0, stream>>>(act2, wfcp, fcb, fco);

    // hyperbolic MLR head
    hyp_mlr<<<(NB * KCLS) / 256, 256, 0, stream>>>(fco, hw, hb, (float*)d_out);
}
